// Encoder_8229157339656
// MI455X (gfx1250) — matrix operand order repaired and checked by reading, not compiled, not run on hardware
//
#include <hip/hip_runtime.h>
#include <hip/hip_bf16.h>
#include <math.h>

typedef __attribute__((ext_vector_type(16))) _Float16 v16h;
typedef __attribute__((ext_vector_type(8)))  _Float16 v8h;
typedef __attribute__((ext_vector_type(8)))  float    v8f;

#define D_  1024
#define F_  4096
#define H_  16
#define DH_ 64
#define S_  2048
#define B_  2
#define T_  (B_*S_)

// ---------------------------------------------------------------------------
// WMMA helpers: fragment loaders per CDNA5 ISA 7.12.2 layouts (wave32).
// A 16x32 f16: lane m=l%16; halves 0..7 -> K = 8*(l/16)+i ; halves 8..15 -> K = 16+8*(l/16)+i
// B 32x16 f16 (stored as Bt row-major [N][K]): lane n=l%16; halves 0..7 -> K = 8*(l/16)+i ; halves 8..15 -> K = 16+8*(l/16)+i
// C/D 16x16 f32: vgpr r, lane l: m = r + 8*(l/16), n = l%16
// ---------------------------------------------------------------------------
__device__ __forceinline__ v16h load_a_frag(const _Float16* __restrict__ base, int ldk) {
    int lane = threadIdx.x & 31;
    int m  = lane & 15;
    int kb = (lane >> 4) * 8;
    v8h lo = *(const v8h*)(base + (size_t)m * ldk + kb);
    v8h hi = *(const v8h*)(base + (size_t)m * ldk + kb + 16);
    v16h a;
#pragma unroll
    for (int i = 0; i < 8; ++i) { a[i] = lo[i]; a[8 + i] = hi[i]; }
    return a;
}

__device__ __forceinline__ v16h load_b_frag(const _Float16* __restrict__ base, int ldk) {
    int lane = threadIdx.x & 31;
    int n  = lane & 15;
    int kb = (lane >> 4) * 8;
    v8h lo = *(const v8h*)(base + (size_t)n * ldk + kb);
    v8h hi = *(const v8h*)(base + (size_t)n * ldk + kb + 16);
    v16h b;
#pragma unroll
    for (int i = 0; i < 8; ++i) { b[i] = lo[i]; b[8 + i] = hi[i]; }
    return b;
}

__device__ __forceinline__ v8f wmma32(v16h a, v16h b, v8f c) {
    return __builtin_amdgcn_wmma_f32_16x16x32_f16(false, a, false, b, (short)0, c, false, false);
}

// DPP16 cross-lane max over the 16-lane half-wave group (VALU pipe, no LDS).
// Steps: quad_perm xor1 (0xB1), quad_perm xor2 (0x4E), row_half_mirror xor7
// (0x141), row_ror:8 xor8 (0x128). Union of coverage = all 16 lanes.
template <int CTRL>
__device__ __forceinline__ float fmax_dpp(float v) {
    int p = __builtin_amdgcn_update_dpp(0, __float_as_int(v), CTRL, 0xF, 0xF, true);
    return fmaxf(v, __int_as_float(p));
}
__device__ __forceinline__ float rowmax16(float v) {
    v = fmax_dpp<0xB1>(v);
    v = fmax_dpp<0x4E>(v);
    v = fmax_dpp<0x141>(v);
    v = fmax_dpp<0x128>(v);
    return v;
}

// ---------------------------------------------------------------------------
// LayerNorm over D=1024, one 256-thread block per token, fp16 output.
// ---------------------------------------------------------------------------
__global__ __launch_bounds__(256) void ln_k(const float* __restrict__ x,
                                            const float* __restrict__ g,
                                            const float* __restrict__ b,
                                            _Float16* __restrict__ out) {
    int tok = blockIdx.x;
    int t = threadIdx.x;
    const float4* x4 = (const float4*)(x + (size_t)tok * D_);
    float4 v = x4[t];
    float s  = v.x + v.y + v.z + v.w;
    float s2 = v.x * v.x + v.y * v.y + v.z * v.z + v.w * v.w;
#pragma unroll
    for (int mask = 1; mask < 32; mask <<= 1) {
        s  += __shfl_xor(s,  mask);
        s2 += __shfl_xor(s2, mask);
    }
    __shared__ float ss[8], ss2[8];
    int wave = t >> 5, lane = t & 31;
    if (lane == 0) { ss[wave] = s; ss2[wave] = s2; }
    __syncthreads();
    float S = 0.f, S2 = 0.f;
#pragma unroll
    for (int i = 0; i < 8; ++i) { S += ss[i]; S2 += ss2[i]; }
    float mean = S * (1.0f / D_);
    float var  = S2 * (1.0f / D_) - mean * mean;
    float rstd = rsqrtf(var + 1e-5f);
    _Float16* o = out + (size_t)tok * D_ + t * 4;
    const float* gp = g + t * 4;
    const float* bp = b + t * 4;
    o[0] = (_Float16)((v.x - mean) * rstd * gp[0] + bp[0]);
    o[1] = (_Float16)((v.y - mean) * rstd * gp[1] + bp[1]);
    o[2] = (_Float16)((v.z - mean) * rstd * gp[2] + bp[2]);
    o[3] = (_Float16)((v.w - mean) * rstd * gp[3] + bp[3]);
}

// ---------------------------------------------------------------------------
// Weight conversions (fp32 -> fp16, Bt layout [N][K]).
// ---------------------------------------------------------------------------
__global__ __launch_bounds__(256) void cvt_qkvw_k(const float* __restrict__ Wq,
                                                  const float* __restrict__ Wk,
                                                  const float* __restrict__ Wv,
                                                  _Float16* __restrict__ out) {
    int idx = blockIdx.x * 256 + threadIdx.x;       // enumerates out[((o*H+h)*DH+dh)*D + d]
    int d    = idx & (D_ - 1);
    int rest = idx >> 10;
    int dh   = rest & (DH_ - 1);
    rest >>= 6;
    int h = rest & (H_ - 1);
    int o = rest >> 4;
    const float* W = (o == 0) ? Wq : ((o == 1) ? Wk : Wv);
    out[idx] = (_Float16)W[(size_t)h * D_ * DH_ + (size_t)d * DH_ + dh];
}

__global__ __launch_bounds__(256) void cvt_T_k(const float* __restrict__ in,
                                               _Float16* __restrict__ out,
                                               int R, int C) {
    int idx = blockIdx.x * 256 + threadIdx.x;
    if (idx >= R * C) return;
    int r = idx / C;
    int c = idx - r * C;
    out[(size_t)c * R + r] = (_Float16)in[idx];
}

// ---------------------------------------------------------------------------
// WMMA GEMM: C[M,Ntot] = A[M,K](f16, row-major) * Bt[Ntot,K](f16, row-major)^T
// block = 256 threads = 8 waves; tile M=128 (16 rows/wave), N=64 (4 frags/wave).
// Weights (Bt) stay resident in the 192 MB L2, so B-operands stream from
// global without LDS staging.
// MODE 0: QKV epilogue (bias, q*0.125, scatter q/k/v^T)
// MODE 1: fp32 out = res + C + bias
// MODE 2: fp16 out = gelu(C + bias)
// ---------------------------------------------------------------------------
template <int MODE>
__global__ __launch_bounds__(256) void gemm_wmma_k(
    const _Float16* __restrict__ A, const _Float16* __restrict__ Bt,
    int K, int Ntot,
    const float* __restrict__ bias, const float* __restrict__ res,
    float* __restrict__ outf, _Float16* __restrict__ outh,
    _Float16* __restrict__ q16, _Float16* __restrict__ k16, _Float16* __restrict__ vt,
    const float* __restrict__ bq, const float* __restrict__ bk, const float* __restrict__ bv) {
    int wave = threadIdx.x >> 5, lane = threadIdx.x & 31;
    int m0 = blockIdx.y * 128 + wave * 16;
    int n0 = blockIdx.x * 64;
    v8f z8 = {0.f, 0.f, 0.f, 0.f, 0.f, 0.f, 0.f, 0.f};
    v8f c[4] = {z8, z8, z8, z8};

    for (int k0 = 0; k0 < K; k0 += 32) {
        v16h a = load_a_frag(A + (size_t)m0 * K + k0, K);
#pragma unroll
        for (int t = 0; t < 4; ++t) {
            v16h b = load_b_frag(Bt + (size_t)(n0 + t * 16) * K + k0, K);
            c[t] = wmma32(a, b, c[t]);
        }
    }

    int g = lane >> 4, nn = lane & 15;
#pragma unroll
    for (int t = 0; t < 4; ++t) {
#pragma unroll
        for (int r = 0; r < 8; ++r) {
            int m = m0 + r + 8 * g;
            int n = n0 + t * 16 + nn;
            float v = c[t][r];
            if (MODE == 0) {
                int o   = n >> 10;
                int rem = n & 1023;
                int h   = rem >> 6;
                int dh  = rem & 63;
                int tb  = m >> 11;            // batch
                int s   = m & (S_ - 1);
                int bh  = tb * H_ + h;
                const float* bb = (o == 0) ? bq : ((o == 1) ? bk : bv);
                float val = v + bb[h * DH_ + dh];
                if (o == 0) {
                    q16[((size_t)bh * S_ + s) * DH_ + dh] = (_Float16)(val * 0.125f);
                } else if (o == 1) {
                    k16[((size_t)bh * S_ + s) * DH_ + dh] = (_Float16)val;
                } else {
                    vt[((size_t)bh * DH_ + dh) * S_ + s] = (_Float16)val;
                }
            } else if (MODE == 1) {
                size_t ix = (size_t)m * Ntot + n;
                outf[ix] = res[ix] + v + bias[n];
            } else {
                float u = v + bias[n];
                outh[(size_t)m * Ntot + n] =
                    (_Float16)(0.5f * u * (1.0f + erff(u * 0.70710678118654752f)));
            }
        }
    }
}

// ---------------------------------------------------------------------------
// Fused flash attention. grid = (S/128, B*H); 8 waves/block; each wave owns a
// 16-row q tile and streams kv in chunks of 32 with online softmax.
// q already scaled by 1/sqrt(DH). k: [bh,S,DH]. v: transposed [bh,DH,S].
// Row-max via DPP16 (VALU); row-sum via ones-column WMMA (matrix pipe): the
// 5th accumulator ol = P x 1 carries the softmax denominator with the same
// alpha-rescaling as O.
// ---------------------------------------------------------------------------
__global__ __launch_bounds__(256) void attn_k(const _Float16* __restrict__ q16,
                                              const _Float16* __restrict__ k16,
                                              const _Float16* __restrict__ vt,
                                              _Float16* __restrict__ ao) {
    __shared__ alignas(16) _Float16 pbuf[8][16][32];
    int wave = threadIdx.x >> 5, lane = threadIdx.x & 31;
    int bh = blockIdx.y;
    int s0 = blockIdx.x * 128 + wave * 16;
    int g = lane >> 4, nn = lane & 15;

    const _Float16* qb = q16 + ((size_t)bh * S_ + s0) * DH_;
    v16h aq0 = load_a_frag(qb, DH_);        // K = 0..31
    v16h aq1 = load_a_frag(qb + 32, DH_);   // K = 32..63

    v16h ones;
#pragma unroll
    for (int i = 0; i < 16; ++i) ones[i] = (_Float16)1.0f;

    v8f z8 = {0.f, 0.f, 0.f, 0.f, 0.f, 0.f, 0.f, 0.f};
    v8f o_[4] = {z8, z8, z8, z8};
    v8f ol = z8;                  // softmax denominator accumulator
    float mi[8];
#pragma unroll
    for (int r = 0; r < 8; ++r) mi[r] = -1e30f;

    const _Float16* kb_base = k16 + (size_t)bh * S_ * DH_;
    const _Float16* vb_base = vt + (size_t)bh * DH_ * S_;

    for (int j = 0; j < S_; j += 32) {
        // ---- scores for kv columns [j, j+32): two 16x16 f32 tiles ----
        v16h b0a = load_b_frag(kb_base + (size_t)j * DH_, DH_);
        v16h b0b = load_b_frag(kb_base + (size_t)j * DH_ + 32, DH_);
        v8f c0 = z8;
        c0 = wmma32(aq0, b0a, c0);
        c0 = wmma32(aq1, b0b, c0);
        v16h b1a = load_b_frag(kb_base + (size_t)(j + 16) * DH_, DH_);
        v16h b1b = load_b_frag(kb_base + (size_t)(j + 16) * DH_ + 32, DH_);
        v8f c1 = z8;
        c1 = wmma32(aq0, b1a, c1);
        c1 = wmma32(aq1, b1b, c1);

        // ---- online softmax: max via DPP16, exp on VALU ----
        float al[8];
#pragma unroll
        for (int r = 0; r < 8; ++r) {
            float vmax = rowmax16(fmaxf(c0[r], c1[r]));
            float mn = fmaxf(mi[r], vmax);
            al[r] = __expf(mi[r] - mn);
            c0[r] = __expf(c0[r] - mn);
            c1[r] = __expf(c1[r] - mn);
            mi[r] = mn;
        }
#pragma unroll
        for (int r = 0; r < 8; ++r) {
            o_[0][r] *= al[r]; o_[1][r] *= al[r];
            o_[2][r] *= al[r]; o_[3][r] *= al[r];
            ol[r]    *= al[r];
        }

        // ---- P (C-layout) -> LDS -> A-fragment relayout ----
        __syncthreads();   // WAR guard vs previous iteration's loads
#pragma unroll
        for (int r = 0; r < 8; ++r) {
            pbuf[wave][r + 8 * g][nn]      = (_Float16)c0[r];
            pbuf[wave][r + 8 * g][16 + nn] = (_Float16)c1[r];
        }
        __syncthreads();   // RAW guard
        v16h ap;
        {
            int kb = 8 * g;
            v8h lo = *(const v8h*)&pbuf[wave][nn][kb];
            v8h hi = *(const v8h*)&pbuf[wave][nn][16 + kb];
#pragma unroll
            for (int i = 0; i < 8; ++i) { ap[i] = lo[i]; ap[8 + i] = hi[i]; }
        }

        // ---- O += P*V ; l += P*1 ----
#pragma unroll
        for (int t = 0; t < 4; ++t) {
            v16h bv = load_b_frag(vb_base + (size_t)(t * 16) * S_ + j, S_);
            o_[t] = wmma32(ap, bv, o_[t]);
        }
        ol = wmma32(ap, ones, ol);
    }

    // ---- normalize and scatter to [B,S,D] with head offset ----
    int b = bh >> 4;
    int h = bh & 15;
#pragma unroll
    for (int r = 0; r < 8; ++r) {
        float inv = 1.0f / ol[r];
        int s = s0 + r + 8 * g;
        size_t tok = (size_t)(b * S_ + s);
#pragma unroll
        for (int t = 0; t < 4; ++t) {
            ao[tok * D_ + h * DH_ + t * 16 + nn] = (_Float16)(o_[t][r] * inv);
        }
    }
}

// ---------------------------------------------------------------------------
// Orchestration
// ---------------------------------------------------------------------------
extern "C" void kernel_launch(void* const* d_in, const int* in_sizes, int n_in,
                              void* d_out, int out_size, void* d_ws, size_t ws_size,
                              hipStream_t stream) {
    (void)in_sizes; (void)n_in; (void)out_size; (void)ws_size;
    const float* x    = (const float*)d_in[0];
    const float* ln1g = (const float*)d_in[1];
    const float* ln1b = (const float*)d_in[2];
    const float* Wq   = (const float*)d_in[3];
    const float* bq   = (const float*)d_in[4];
    const float* Wk   = (const float*)d_in[5];
    const float* bk   = (const float*)d_in[6];
    const float* Wv   = (const float*)d_in[7];
    const float* bv   = (const float*)d_in[8];
    const float* Wo   = (const float*)d_in[9];
    const float* bo   = (const float*)d_in[10];
    const float* ln2g = (const float*)d_in[11];
    const float* ln2b = (const float*)d_in[12];
    const float* W1   = (const float*)d_in[13];
    const float* b1   = (const float*)d_in[14];
    const float* W2   = (const float*)d_in[15];
    const float* b2   = (const float*)d_in[16];
    float* out = (float*)d_out;

    char* ws = (char*)d_ws;
    const size_t MB = 1ull << 20;
    _Float16* h16  = (_Float16*)(ws + 0 * MB);   // 8 MB  (reused as attn-out)
    _Float16* ao16 = h16;
    _Float16* qkvw = (_Float16*)(ws + 8 * MB);   // 6 MB
    _Float16* q16  = (_Float16*)(ws + 16 * MB);  // 8 MB  (reused as Wo^T)
    _Float16* k16  = (_Float16*)(ws + 24 * MB);  // 8 MB  (reused as W1^T)
    _Float16* vt16 = (_Float16*)(ws + 32 * MB);  // 8 MB  (reused as W2^T)
    float*    x1   = (float*)   (ws + 40 * MB);  // 16 MB
    _Float16* h2   = (_Float16*)(ws + 56 * MB);  // 8 MB
    _Float16* g16  = (_Float16*)(ws + 64 * MB);  // 32 MB
    _Float16* wot = q16;
    _Float16* w1t = k16;
    _Float16* w2t = vt16;

    // 1) LN1
    ln_k<<<T_, 256, 0, stream>>>(x, ln1g, ln1b, h16);
    // 2) pack QKV weights fp16 Bt-layout
    cvt_qkvw_k<<<(3 * H_ * DH_ * D_) / 256, 256, 0, stream>>>(Wq, Wk, Wv, qkvw);
    // 3) fused QKV projection (q scaled by 1/8, v stored transposed)
    gemm_wmma_k<0><<<dim3(3 * D_ / 64, T_ / 128), 256, 0, stream>>>(
        h16, qkvw, D_, 3 * D_, nullptr, nullptr, nullptr, nullptr,
        q16, k16, vt16, bq, bk, bv);
    // 4) flash attention -> ao16 (overwrites h16 region)
    attn_k<<<dim3(S_ / 128, B_ * H_), 256, 0, stream>>>(q16, k16, vt16, ao16);
    // 5) Wo + residual -> x1 (fp32)
    cvt_T_k<<<(D_ * D_) / 256, 256, 0, stream>>>(Wo, wot, D_, D_);
    gemm_wmma_k<1><<<dim3(D_ / 64, T_ / 128), 256, 0, stream>>>(
        ao16, wot, D_, D_, bo, x, x1, nullptr,
        nullptr, nullptr, nullptr, nullptr, nullptr, nullptr);
    // 6) LN2
    ln_k<<<T_, 256, 0, stream>>>(x1, ln2g, ln2b, h2);
    // 7) FFN up + exact GELU -> g16
    cvt_T_k<<<(D_ * F_) / 256, 256, 0, stream>>>(W1, w1t, D_, F_);
    gemm_wmma_k<2><<<dim3(F_ / 64, T_ / 128), 256, 0, stream>>>(
        h2, w1t, D_, F_, b1, nullptr, nullptr, g16,
        nullptr, nullptr, nullptr, nullptr, nullptr, nullptr);
    // 8) FFN down + residual -> out
    cvt_T_k<<<(F_ * D_) / 256, 256, 0, stream>>>(W2, w2t, F_, D_);
    gemm_wmma_k<1><<<dim3(D_ / 64, T_ / 128), 256, 0, stream>>>(
        g16, w2t, F_, D_, b2, x1, out, nullptr,
        nullptr, nullptr, nullptr, nullptr, nullptr, nullptr);
}
